// KANLinear_6287832121481
// MI455X (gfx1250) — matrix operand order repaired and checked by reading, not compiled, not run on hardware
//
#include <hip/hip_runtime.h>
#include <hip/hip_bf16.h>
#include <math.h>

// ---------------------------------------------------------------------------
// KAN linear layer as one f16-in/f32-acc WMMA GEMM on gfx1250.
//   M = 16384, N = 256, K = 256*8 (spline basis) + 256 (silu base) = 2304
//   ~19.3 GFLOP vs ~34 MB moved -> ~570 FLOP/byte: compute bound, so use the
//   highest-throughput fp path with enough precision: v_wmma_f32_16x16x32_f16.
// Wave tiling: 2 M-tiles x 4 N-tiles per wave -> each B fragment feeds two
// WMMAs, and B fragments are double-buffered to hide ds_load latency.
// ---------------------------------------------------------------------------

typedef __attribute__((ext_vector_type(16))) _Float16 v16h;
typedef __attribute__((ext_vector_type(8)))  _Float16 v8h;
typedef __attribute__((ext_vector_type(8)))  float    v8f;

#define K_CHUNKS   72      // 64 spline chunks (4 features x 8 coeffs) + 8 base chunks
#define BM         64      // M rows per block
#define APAD       40      // padded row length (halfs) for LDS tiles
#define NFEAT      256
#define NOUT       256

// ---------------------------------------------------------------------------
// Kernel 1: pack weight (256 i, 256 o, 9 q) f32 -> f16 B-matrix, chunk-tiled:
//   wt[c][o][kk], c<64: (i = 4c + kk/8, q = kk%8), c>=64: base column (q=8).
// One-time ~1.1MB; makes the GEMM's B fill a perfectly coalesced 64B/thread copy.
// ---------------------------------------------------------------------------
__global__ __launch_bounds__(256) void kan_weight_pack(
    const float* __restrict__ weight, _Float16* __restrict__ wt) {
  const int c = blockIdx.x;     // 0..71
  const int o = threadIdx.x;    // 0..255
  _Float16 tmp[32];
  if (c < 64) {
    const int i0 = c * 4;
#pragma unroll
    for (int kk = 0; kk < 32; ++kk) {
      const int i = i0 + (kk >> 3);
      const int q = kk & 7;
      tmp[kk] = (_Float16)weight[((size_t)i * NOUT + o) * 9 + q];
    }
  } else {
    const int i0 = (c - 64) * 32;
#pragma unroll
    for (int kk = 0; kk < 32; ++kk)
      tmp[kk] = (_Float16)weight[((size_t)(i0 + kk) * NOUT + o) * 9 + 8];
  }
  v8h* dst = (v8h*)(wt + ((size_t)c * NOUT + o) * 32);
#pragma unroll
  for (int j = 0; j < 4; ++j) dst[j] = *(v8h*)&tmp[j * 8];
}

// Cubic B-spline basis (s=3, g=5, grid [-1,1], h=0.4), matching the reference:
// only 4 of the 8 coeffs are nonzero; closed-form from the 4x4 basis matrix.
__device__ __forceinline__ v8h kan_basis8(float v) {
  const bool inr = (v >= -2.2f) && (v < 2.2f);
  int idx = (int)floorf((v + 2.2f) * 2.5f);
  idx = idx < 0 ? 0 : (idx > 10 ? 10 : idx);
  const float u  = (v - (0.4f * (float)idx - 1.0f)) * 2.5f;
  const float u2 = u * u, u3 = u2 * u;
  const float k6 = 1.0f / 6.0f;
  const float p0 = (1.0f - 3.0f * u + 3.0f * u2 - u3) * k6;
  const float p1 = (4.0f - 6.0f * u2 + 3.0f * u3) * k6;
  const float p2 = (1.0f + 3.0f * u + 3.0f * u2 - 3.0f * u3) * k6;
  const float p3 = u3 * k6;
  v8h out;
#pragma unroll
  for (int q = 0; q < 8; ++q) {
    const int j = q + 3 - idx;
    float val = 0.0f;
    val = (j == 0) ? p0 : val;
    val = (j == 1) ? p1 : val;
    val = (j == 2) ? p2 : val;
    val = (j == 3) ? p3 : val;
    out[q] = (_Float16)(inr ? val : 0.0f);
  }
  return out;
}

__device__ __forceinline__ float silu_f(float v) {
  return v / (1.0f + __expf(-v));
}

// Build a 16x32 f16 A/B operand (v16h) from two 16B LDS rows.
__device__ __forceinline__ v16h frag16(const _Float16* p0, const _Float16* p1) {
  const v8h lo = *(const v8h*)p0;
  const v8h hi = *(const v8h*)p1;
  v16h f;
#pragma unroll
  for (int e = 0; e < 8; ++e) { f[e] = lo[e]; f[e + 8] = hi[e]; }
  return f;
}

// ---------------------------------------------------------------------------
// Kernel 2: fused activation-build + GEMM.
// Block: 256 threads (8 wave32s), 64x256 output tile.
// Wave w: M-tile pair mth = w&1 (rows mth*32 + {0,16}), N quarter nh = w>>1
// (cols nh*64 + t*16, t=0..3) -> acc[2][4] v8f tiles.
// ---------------------------------------------------------------------------
__global__ __launch_bounds__(256) void kan_gemm(
    const float* __restrict__ x, const _Float16* __restrict__ wt,
    float* __restrict__ out) {
  __shared__ __align__(16) _Float16 sA[BM][APAD];    //  64x32 f16 A tile (padded)
  __shared__ __align__(16) _Float16 sB[NOUT][APAD];  // 256x32 f16 B tile (padded)

  const int tid  = threadIdx.x;
  const int m0   = blockIdx.x * BM;
  const int lane = tid & 31;
  const int wv   = tid >> 5;
  const int mth  = wv & 1;            // which 32-row half (2 M-tiles)
  const int nh   = wv >> 1;           // which 64-col quarter (4 N-tiles)
  const int m    = lane & 15;
  const int hi   = lane >> 4;         // lane group (K-split select)
  const int klA  = hi ? 16 : 0;       // A: lanes 0-15 hold K 0-15; lanes 16-31: K 16-31
  const int klB  = hi ? 16 : 0;       // B: lanes 0-15 hold K 0-15; lanes 16-31: K 16-31

  // A-tile fill assignment: thread -> (row, sub) of the 64x32 tile
  const int frow = tid & 63;
  const int fsub = tid >> 6;          // 0..3

  v8f acc[2][4];
#pragma unroll
  for (int j = 0; j < 2; ++j)
#pragma unroll
    for (int t = 0; t < 4; ++t)
      acc[j][t] = (v8f){0.f, 0.f, 0.f, 0.f, 0.f, 0.f, 0.f, 0.f};

  for (int c = 0; c < K_CHUNKS; ++c) {
    // ---- stage 1: gather fill data into registers (no LDS yet) -------------
    const v8h* __restrict__ bsrc = (const v8h*)(wt + ((size_t)c * NOUT + tid) * 32);
    const v8h b0 = bsrc[0], b1 = bsrc[1], b2 = bsrc[2], b3 = bsrc[3];
    if (c + 1 < K_CHUNKS)  // global_prefetch_b8 for next weight chunk (16KB/block)
      __builtin_prefetch(wt + ((size_t)(c + 1) * NOUT) * 32 + (size_t)tid * 32, 0, 1);

    v8h afrag;
    if (c < 64) {
      // spline chunk: k = i*8+q, i in [4c, 4c+4); one x value -> 8 basis coeffs
      const int i = c * 4 + fsub;
      const float v = x[(size_t)(m0 + frow) * NFEAT + i];
      afrag = kan_basis8(v);
    } else {
      // base chunk: k = 2048 + i, i in [32(c-64), +32); 8 silu values
      const int i0 = (c - 64) * 32 + fsub * 8;
      const float* xp = &x[(size_t)(m0 + frow) * NFEAT + i0];
      const float4 xa = *(const float4*)xp;
      const float4 xb = *(const float4*)(xp + 4);
      afrag[0] = (_Float16)silu_f(xa.x); afrag[1] = (_Float16)silu_f(xa.y);
      afrag[2] = (_Float16)silu_f(xa.z); afrag[3] = (_Float16)silu_f(xa.w);
      afrag[4] = (_Float16)silu_f(xb.x); afrag[5] = (_Float16)silu_f(xb.y);
      afrag[6] = (_Float16)silu_f(xb.z); afrag[7] = (_Float16)silu_f(xb.w);
    }

    __syncthreads();  // all waves finished reading the previous chunk's tiles

    // ---- stage 2: write LDS tiles -----------------------------------------
    *(v8h*)&sA[frow][fsub * 8] = afrag;
    v8h* brow = (v8h*)&sB[tid][0];
    brow[0] = b0; brow[1] = b1; brow[2] = b2; brow[3] = b3;

    __syncthreads();  // tiles visible to all waves

    // ---- stage 3: WMMA (2 A-tiles reuse each B fragment; B double-buffered)
    const v16h a0 = frag16(&sA[mth * 32 + m][klA],      &sA[mth * 32 + m][klA + 8]);
    const v16h a1 = frag16(&sA[mth * 32 + 16 + m][klA], &sA[mth * 32 + 16 + m][klA + 8]);

    v16h bcur = frag16(&sB[nh * 64 + m][klB], &sB[nh * 64 + m][klB + 8]);
#pragma unroll
    for (int t = 0; t < 4; ++t) {
      v16h bnext = bcur;
      if (t < 3) {
        const int n = nh * 64 + (t + 1) * 16 + m;
        bnext = frag16(&sB[n][klB], &sB[n][klB + 8]);
      }
      acc[0][t] = __builtin_amdgcn_wmma_f32_16x16x32_f16(
          false, a0, false, bcur, (short)0, acc[0][t], false, false);
      acc[1][t] = __builtin_amdgcn_wmma_f32_16x16x32_f16(
          false, a1, false, bcur, (short)0, acc[1][t], false, false);
      bcur = bnext;
    }
  }

  // ---- epilogue: C/D layout -> global. VGPR r: row = r + 8*hi, col = lane&15.
#pragma unroll
  for (int j = 0; j < 2; ++j)
#pragma unroll
    for (int t = 0; t < 4; ++t) {
      const int n = nh * 64 + t * 16 + m;
#pragma unroll
      for (int r = 0; r < 8; ++r) {
        const int row = m0 + mth * 32 + j * 16 + hi * 8 + r;
        out[(size_t)row * NOUT + n] = acc[j][t][r];
      }
    }
}

// ---------------------------------------------------------------------------
extern "C" void kernel_launch(void* const* d_in, const int* in_sizes, int n_in,
                              void* d_out, int out_size, void* d_ws, size_t ws_size,
                              hipStream_t stream) {
  const float* x      = (const float*)d_in[0];   // (16384, 256) f32
  const float* weight = (const float*)d_in[1];   // (256, 256, 9) f32
  float*       out    = (float*)d_out;           // (16384, 256) f32
  _Float16*    wt     = (_Float16*)d_ws;         // 72*256*32 f16 = ~1.13 MB

  kan_weight_pack<<<dim3(K_CHUNKS), dim3(256), 0, stream>>>(weight, wt);
  kan_gemm<<<dim3(16384 / BM), dim3(256), 0, stream>>>(x, wt, out);
}
